// DyGCAPlugin_74972949119143
// MI455X (gfx1250) — compile-verified, matrix operand order checked by reading, not run on hardware
//
#include <hip/hip_runtime.h>
#include <hip/hip_bf16.h>

#define S_LEN 1024
#define HDIM  768
#define KCH   32
#define MSEL  8
#define GQ    32
#define VOCAB 32000

typedef __attribute__((ext_vector_type(16))) __bf16 v16bf;
typedef __attribute__((ext_vector_type(8)))  float  v8f;

union Frag16 {
  v16bf v;
  unsigned short u[16];
  uint4 q[2];
};

__device__ __forceinline__ unsigned short f2bf(float f) {
  unsigned int u = __float_as_uint(f);
  unsigned int r = u + 0x7FFFu + ((u >> 16) & 1u);
  return (unsigned short)(r >> 16);
}

__device__ __forceinline__ float softplusf(float x) {
  return (x > 20.f) ? x : log1pf(__expf(x));
}

// ---------------------------------------------------------------------------
// Tables: log position grid + Gauss-Legendre nodes/weights (Newton on P_32)
// ---------------------------------------------------------------------------
__global__ __launch_bounds__(256) void k_tables(float* lx, float* l1p,
                                                float* lnn, float* l1pn, float* glw) {
  int tid = threadIdx.x;
  for (int j = tid; j < S_LEN; j += 256) {
    float x = (j + 0.5f) / (float)S_LEN;
    lx[j]  = __logf(x);
    l1p[j] = log1pf(-x);
  }
  if (tid < GQ) {
    float x = cosf(3.14159265358979f * (tid + 0.75f) / (GQ + 0.5f));
    float dp = 1.f;
    for (int it = 0; it < 64; ++it) {
      float p0 = 1.f, p1 = x;
      for (int n = 2; n <= GQ; ++n) {
        float p2 = ((2.f * n - 1.f) * x * p1 - (n - 1.f) * p0) / (float)n;
        p0 = p1; p1 = p2;
      }
      dp = GQ * (x * p1 - p0) / (x * x - 1.f);
      x -= p1 / dp;
    }
    float w = 2.f / ((1.f - x * x) * dp * dp);
    float node = 0.5f * (x + 1.f);
    lnn[tid]  = __logf(node);
    l1pn[tid] = log1pf(-node);
    glw[tid]  = 0.5f * w;
  }
}

// ---------------------------------------------------------------------------
// Embedding gather -> transposed bf16 base: baseT[h][t]
// ---------------------------------------------------------------------------
__global__ __launch_bounds__(256) void k_gather(const int* ids, const float* embed,
                                                unsigned short* baseT) {
  int idx = blockIdx.x * 256 + threadIdx.x;  // H*S
  int h = idx / S_LEN;
  int t = idx % S_LEN;
  baseT[idx] = f2bf(embed[(size_t)ids[t] * HDIM + h]);
}

__global__ __launch_bounds__(256) void k_cvt(const float* in, unsigned short* out, int n) {
  int i = blockIdx.x * 256 + threadIdx.x;
  if (i < n) out[i] = f2bf(in[i]);
}

// ---------------------------------------------------------------------------
// focus/imp projections (shifted by one step) -> alpha-1, beta-1, lgamma const, imp
// ---------------------------------------------------------------------------
__global__ __launch_bounds__(256) void k_proj(const float* h_pen,
                                              const float* focus_w, const float* focus_b,
                                              const float* imp_w, const float* imp_b,
                                              float* am1, float* bm1, float* cc, float* impv) {
  int idx = blockIdx.x * 256 + threadIdx.x;  // S*K
  int t = idx / KCH, k = idx % KCH;
  float d0 = 0.f, d1 = 0.f, di = 0.f, iv = 0.f;
  if (t > 0) {
    const float4* hp = (const float4*)(h_pen + (size_t)(t - 1) * HDIM);
    const float4* w0 = (const float4*)(focus_w + (size_t)(2 * k) * HDIM);
    const float4* w1 = (const float4*)(focus_w + (size_t)(2 * k + 1) * HDIM);
    const float4* wi = (const float4*)(imp_w + (size_t)k * HDIM);
    for (int i = 0; i < HDIM / 4; ++i) {
      float4 h4 = hp[i], a4 = w0[i], b4 = w1[i], c4 = wi[i];
      d0 += h4.x * a4.x + h4.y * a4.y + h4.z * a4.z + h4.w * a4.w;
      d1 += h4.x * b4.x + h4.y * b4.y + h4.z * b4.z + h4.w * b4.w;
      di += h4.x * c4.x + h4.y * c4.y + h4.z * c4.z + h4.w * c4.w;
    }
    d0 += focus_b[2 * k]; d1 += focus_b[2 * k + 1]; di += imp_b[k];
    iv = softplusf(di);
  }
  float a = softplusf(d0) + 1e-4f;
  float b = softplusf(d1) + 1e-4f;
  am1[idx] = a - 1.f;
  bm1[idx] = b - 1.f;
  cc[idx]  = lgammaf(a) + lgammaf(b) - lgammaf(a + b);
  impv[idx] = iv;
}

// ---------------------------------------------------------------------------
// Beta pdf at quadrature nodes, plus log(p + eps)
// ---------------------------------------------------------------------------
__global__ __launch_bounds__(256) void k_pdfg(const float* am1, const float* bm1,
                                              const float* cc,
                                              const float* lnn, const float* l1pn,
                                              float* pdfg, float* lp) {
  int idx = blockIdx.x * 256 + threadIdx.x;  // S*K
  float a = am1[idx], b = bm1[idx], c = cc[idx];
  for (int g = 0; g < GQ; ++g) {
    float e = __expf(a * lnn[g] + b * l1pn[g] - c);
    pdfg[(size_t)idx * GQ + g] = e;
    lp[(size_t)idx * GQ + g]   = __logf(e + 1e-9f);
  }
}

// ---------------------------------------------------------------------------
// top-M selection (ties -> lower index, matching jax top_k), slot map per k
// ---------------------------------------------------------------------------
__global__ __launch_bounds__(256) void k_topk(const float* impv, int* selmap) {
  int t = blockIdx.x * 256 + threadIdx.x;
  if (t >= S_LEN) return;
  float v[KCH]; int sm[KCH];
  for (int k = 0; k < KCH; ++k) { v[k] = impv[t * KCH + k]; sm[k] = -1; }
  for (int s = 0; s < MSEL; ++s) {
    int bi = 0; float bv = v[0];
    for (int k = 1; k < KCH; ++k) if (v[k] > bv) { bv = v[k]; bi = k; }
    sm[bi] = s; v[bi] = -1e30f;
  }
  for (int k = 0; k < KCH; ++k) selmap[t * KCH + k] = sm[k];
}

// ---------------------------------------------------------------------------
// JS divergence per t (sum over K*(K-1)/2 pairs x G nodes) -> jsum[t]
// ---------------------------------------------------------------------------
__global__ __launch_bounds__(256) void k_js(const float* pdfg, const float* lp,
                                            const float* glw, float* jsum) {
  int t = blockIdx.x, tid = threadIdx.x;
  __shared__ float sp[KCH * GQ], sl[KCH * GQ], sw[GQ];
  __shared__ float red[256];
  for (int i = tid; i < KCH * GQ; i += 256) {
    sp[i] = pdfg[(size_t)t * KCH * GQ + i];
    sl[i] = lp[(size_t)t * KCH * GQ + i];
  }
  if (tid < GQ) sw[tid] = glw[tid];
  __syncthreads();
  float acc = 0.f;
  for (int p = tid; p < KCH * (KCH - 1) / 2; p += 256) {
    int i = 0, rem = p;
    while (rem >= KCH - 1 - i) { rem -= KCH - 1 - i; ++i; }
    int j = i + 1 + rem;
    const float* pi = sp + i * GQ; const float* pj = sp + j * GQ;
    const float* li = sl + i * GQ; const float* lj = sl + j * GQ;
    float s = 0.f;
    for (int g = 0; g < GQ; ++g) {
      float m = 0.5f * (pi[g] + pj[g]);
      float lm = __logf(m + 1e-9f);
      s += sw[g] * 0.5f * (pi[g] * (li[g] - lm) + pj[g] * (lj[g] - lm));
    }
    acc += s;
  }
  red[tid] = acc; __syncthreads();
  for (int s = 128; s > 0; s >>= 1) { if (tid < s) red[tid] += red[tid + s]; __syncthreads(); }
  if (tid == 0) jsum[t] = red[0];
}

__global__ __launch_bounds__(256) void k_lossred(const float* jsum, float* out) {
  __shared__ float red[256];
  int tid = threadIdx.x;
  float s = 0.f;
  for (int i = tid; i < S_LEN; i += 256) s += jsum[i];
  red[tid] = s; __syncthreads();
  for (int st = 128; st > 0; st >>= 1) { if (tid < st) red[tid] += red[tid + st]; __syncthreads(); }
  if (tid == 0) out[0] = red[0] * (1.f / 507904.f);  // B*S*K*(K-1)/2
}

// ---------------------------------------------------------------------------
// Fused Beta-attention chunk GEMM:  chunk[t,k,:] = sum_j w[t,k,j]*base[j,:]
// One block per t. Weights computed to LDS per 32-wide j-chunk (bf16 A tiles),
// bf16 WMMA against transposed base, epilogue emits sel (top-M rows) + c_mean.
// ---------------------------------------------------------------------------
__global__ __launch_bounds__(256) void k_chunk(const float* am1, const float* bm1,
                                               const float* cc, const float* impv,
                                               const float* lx, const float* l1p,
                                               const int* selmap,
                                               const unsigned short* baseT,
                                               unsigned short* selbf, float* cmean) {
  int t = blockIdx.x, tid = threadIdx.x;
  int wave = tid >> 5, lane = tid & 31;
  __shared__ float sa[KCH], sb[KCH], sc[KCH], sf[KCH];
  __shared__ int   ssel[KCH];
  __shared__ float red[8][KCH];
  __shared__ float slx[S_LEN], sl1[S_LEN];
  __shared__ unsigned short wlds[2][16][32];

  int nchunks = (t >> 5) + 1;
  int jmax = nchunks * 32;
  for (int j = tid; j < jmax; j += 256) { slx[j] = lx[j]; sl1[j] = l1p[j]; }
  if (tid < KCH) {
    sa[tid] = am1[t * KCH + tid];
    sb[tid] = bm1[t * KCH + tid];
    sc[tid] = cc[t * KCH + tid];
    ssel[tid] = selmap[t * KCH + tid];
  }
  __syncthreads();

  // phase 1: per-row pdf sums over j<=t
  {
    int k = tid & 31, part = tid >> 5;
    float a = sa[k], b = sb[k], c = sc[k], s = 0.f;
    for (int j = part; j <= t; j += 8)
      s += __expf(a * slx[j] + b * sl1[j] - c);
    red[part][k] = s;
  }
  __syncthreads();
  if (tid < KCH) {
    float s = 0.f;
    for (int p = 0; p < 8; ++p) s += red[p][tid];
    sf[tid] = impv[t * KCH + tid] / (s + 1e-9f);
  }
  __syncthreads();

  v8f acc[2][6];
  {
    v8f vz = {};
    for (int a = 0; a < 2; ++a)
      for (int n = 0; n < 6; ++n) acc[a][n] = vz;
  }

  int mrow = lane & 15, hi = lane >> 4;
  int h0 = wave * 96;  // each wave owns 96 h-columns (6 N-tiles)

  for (int jc = 0; jc < nchunks; ++jc) {
    int j0 = jc * 32;
#pragma unroll
    for (int v = 0; v < 4; ++v) {
      int idx = tid * 4 + v;  // 0..1023 -> (k, jj)
      int k = idx >> 5, jj = idx & 31;
      int j = j0 + jj;
      float w = 0.f;
      if (j <= t) w = __expf(sa[k] * slx[j] + sb[k] * sl1[j] - sc[k]) * sf[k];
      wlds[k >> 4][k & 15][jj] = f2bf(w);
    }
    __syncthreads();

    Frag16 a0, a1;
    a0.q[0] = *(const uint4*)&wlds[0][mrow][hi * 8];
    a0.q[1] = *(const uint4*)&wlds[0][mrow][16 + hi * 8];
    a1.q[0] = *(const uint4*)&wlds[1][mrow][hi * 8];
    a1.q[1] = *(const uint4*)&wlds[1][mrow][16 + hi * 8];

#pragma unroll
    for (int nt = 0; nt < 6; ++nt) {
      int h = h0 + nt * 16 + mrow;
      const unsigned short* brow = baseT + (size_t)h * S_LEN + j0;
      Frag16 bfr;
      bfr.q[0] = *(const uint4*)(brow + hi * 8);
      bfr.q[1] = *(const uint4*)(brow + 16 + hi * 8);
      acc[0][nt] = __builtin_amdgcn_wmma_f32_16x16x32_bf16(
          false, a0.v, false, bfr.v, (short)0, acc[0][nt], false, false);
      acc[1][nt] = __builtin_amdgcn_wmma_f32_16x16x32_bf16(
          false, a1.v, false, bfr.v, (short)0, acc[1][nt], false, false);
    }
    __syncthreads();
  }

  // epilogue: scatter selected rows (bf16) and accumulate c_mean
#pragma unroll
  for (int nt = 0; nt < 6; ++nt) {
    int h = h0 + nt * 16 + mrow;  // C-tile: N = lane%16
    float csum = 0.f;
#pragma unroll
    for (int kt = 0; kt < 2; ++kt) {
#pragma unroll
      for (int r = 0; r < 8; ++r) {
        float val = acc[kt][nt][r];
        csum += val;
        int k = kt * 16 + hi * 8 + r;  // C-tile: M = r + 8*(lane/16)
        int slot = ssel[k];
        if (slot >= 0)
          selbf[((size_t)t * MSEL + slot) * HDIM + h] = f2bf(val);
      }
    }
    csum += __shfl_xor(csum, 16, 32);
    if (hi == 0) cmean[(size_t)t * HDIM + h] = csum * (1.f / KCH);
  }
}

// ---------------------------------------------------------------------------
// Generic C = A(bf16) * W(bf16, row-major [cols][768])^T + bias, f32 out.
// Weights are pre-converted to bf16 (one streaming pass) so every fragment is
// two 16-byte global loads; no conversion VALU in the hot loop.
// ---------------------------------------------------------------------------
__global__ __launch_bounds__(256) void k_gemm(const unsigned short* A,
                                              const unsigned short* W,
                                              const float* bias, float* C,
                                              int rows, int cols) {
  int wave = threadIdx.x >> 5, lane = threadIdx.x & 31;
  int r0 = blockIdx.x * 128 + wave * 16;
  int rr = lane & 15, hi = lane >> 4;
  for (int nc = 0; nc < cols; nc += 128) {
    v8f acc[8];
    {
      v8f vz = {};
      for (int n = 0; n < 8; ++n) acc[n] = vz;
    }
    for (int i0 = 0; i0 < HDIM; i0 += 32) {
      Frag16 a;
      const unsigned short* arow = A + (size_t)(r0 + rr) * HDIM + i0;
      a.q[0] = *(const uint4*)(arow + hi * 8);
      a.q[1] = *(const uint4*)(arow + 16 + hi * 8);
#pragma unroll
      for (int nt = 0; nt < 8; ++nt) {
        const unsigned short* wrow = W + (size_t)(nc + nt * 16 + rr) * HDIM + i0;
        __builtin_prefetch(wrow + 128, 0, 1);
        Frag16 b;
        b.q[0] = *(const uint4*)(wrow + hi * 8);
        b.q[1] = *(const uint4*)(wrow + 16 + hi * 8);
        acc[nt] = __builtin_amdgcn_wmma_f32_16x16x32_bf16(
            false, a.v, false, b.v, (short)0, acc[nt], false, false);
      }
    }
#pragma unroll
    for (int nt = 0; nt < 8; ++nt) {
      int o = nc + nt * 16 + rr;
      float bv = bias ? bias[o] : 0.f;
#pragma unroll
      for (int r = 0; r < 8; ++r) {
        int row = r0 + hi * 8 + r;
        C[(size_t)row * cols + o] = acc[nt][r] + bv;
      }
    }
  }
}

// ---------------------------------------------------------------------------
// Attention over M=8 + gate + LayerNorm -> bf16 hidden
// ---------------------------------------------------------------------------
__global__ __launch_bounds__(256) void k_attn(const float* q, const float* kk, const float* vv,
                                              const float* h_pen, const float* h_final,
                                              const float* cmean,
                                              const float* gate_w, const float* gate_b,
                                              const float* ln_g, const float* ln_b,
                                              unsigned short* hidbf) {
  int t = blockIdx.x, tid = threadIdx.x;
  __shared__ float red[256];
  __shared__ float attn[MSEL];
  __shared__ float sG, sMu, sRstd;

  float part[MSEL];
#pragma unroll
  for (int m = 0; m < MSEL; ++m) part[m] = 0.f;
  for (int h = tid; h < HDIM; h += 256) {
    float qh = q[(size_t)t * HDIM + h];
#pragma unroll
    for (int m = 0; m < MSEL; ++m)
      part[m] += qh * kk[((size_t)t * MSEL + m) * HDIM + h];
  }
  for (int m = 0; m < MSEL; ++m) {
    red[tid] = part[m]; __syncthreads();
    for (int s = 128; s > 0; s >>= 1) { if (tid < s) red[tid] += red[tid + s]; __syncthreads(); }
    if (tid == 0) attn[m] = red[0] * 0.036084391824351615f;  // 1/sqrt(768)
    __syncthreads();
  }
  if (tid == 0) {
    float mx = attn[0];
    for (int m = 1; m < MSEL; ++m) mx = fmaxf(mx, attn[m]);
    float se = 0.f;
    for (int m = 0; m < MSEL; ++m) { attn[m] = __expf(attn[m] - mx); se += attn[m]; }
    for (int m = 0; m < MSEL; ++m) attn[m] /= se;
  }
  // gate
  float gp = 0.f;
  for (int h = tid; h < HDIM; h += 256)
    gp += gate_w[h] * h_pen[(size_t)t * HDIM + h]
        + gate_w[HDIM + h] * cmean[(size_t)t * HDIM + h];
  red[tid] = gp; __syncthreads();
  for (int s = 128; s > 0; s >>= 1) { if (tid < s) red[tid] += red[tid + s]; __syncthreads(); }
  if (tid == 0) sG = 1.f / (1.f + __expf(-(red[0] + gate_b[0])));
  __syncthreads();

  float g = sG;
  float hv[3]; int hs[3]; int cnt = 0;
  float sum = 0.f, sumsq = 0.f;
  for (int h = tid; h < HDIM; h += 256) {
    float f = 0.f;
#pragma unroll
    for (int m = 0; m < MSEL; ++m)
      f += attn[m] * vv[((size_t)t * MSEL + m) * HDIM + h];
    float hd = g * f + (1.f - g) * h_final[(size_t)t * HDIM + h];
    hv[cnt] = hd; hs[cnt] = h; ++cnt;
    sum += hd; sumsq += hd * hd;
  }
  red[tid] = sum; __syncthreads();
  for (int s = 128; s > 0; s >>= 1) { if (tid < s) red[tid] += red[tid + s]; __syncthreads(); }
  if (tid == 0) sMu = red[0] / HDIM;
  __syncthreads();
  red[tid] = sumsq; __syncthreads();
  for (int s = 128; s > 0; s >>= 1) { if (tid < s) red[tid] += red[tid + s]; __syncthreads(); }
  if (tid == 0) sRstd = rsqrtf(red[0] / HDIM - sMu * sMu + 1e-5f);
  __syncthreads();
  float mu = sMu, rs = sRstd;
  for (int ii = 0; ii < cnt; ++ii) {
    int h = hs[ii];
    float y = (hv[ii] - mu) * rs * ln_g[h] + ln_b[h];
    hidbf[(size_t)t * HDIM + h] = f2bf(y);
  }
}

// ---------------------------------------------------------------------------
// Host-side orchestration
// ---------------------------------------------------------------------------
extern "C" void kernel_launch(void* const* d_in, const int* in_sizes, int n_in,
                              void* d_out, int out_size, void* d_ws, size_t ws_size,
                              hipStream_t stream) {
  (void)in_sizes; (void)n_in; (void)out_size; (void)ws_size;
  const int*   ids     = (const int*)  d_in[0];
  const float* embed   = (const float*)d_in[1];
  const float* h_pen   = (const float*)d_in[2];
  const float* h_final = (const float*)d_in[3];
  const float* focus_w = (const float*)d_in[4];
  const float* focus_b = (const float*)d_in[5];
  const float* imp_w   = (const float*)d_in[6];
  const float* imp_b   = (const float*)d_in[7];
  const float* q_w     = (const float*)d_in[8];
  const float* q_b     = (const float*)d_in[9];
  const float* k_w     = (const float*)d_in[10];
  const float* k_b     = (const float*)d_in[11];
  const float* v_w     = (const float*)d_in[12];
  const float* v_b     = (const float*)d_in[13];
  const float* gate_w  = (const float*)d_in[14];
  const float* gate_b  = (const float*)d_in[15];
  const float* ln_g    = (const float*)d_in[16];
  const float* ln_b    = (const float*)d_in[17];
  const float* lm_w    = (const float*)d_in[18];

  char* ws = (char*)d_ws;
  size_t off = 0;
  auto alloc = [&](size_t bytes) -> char* {
    char* p = ws + off;
    off = (off + bytes + 255) & ~(size_t)255;
    return p;
  };
  float* lx    = (float*)alloc((size_t)S_LEN * 4);
  float* l1p   = (float*)alloc((size_t)S_LEN * 4);
  float* lnn   = (float*)alloc((size_t)GQ * 4);
  float* l1pn  = (float*)alloc((size_t)GQ * 4);
  float* glw   = (float*)alloc((size_t)GQ * 4);
  unsigned short* baseT = (unsigned short*)alloc((size_t)HDIM * S_LEN * 2);
  float* am1   = (float*)alloc((size_t)S_LEN * KCH * 4);
  float* bm1   = (float*)alloc((size_t)S_LEN * KCH * 4);
  float* cc    = (float*)alloc((size_t)S_LEN * KCH * 4);
  float* impv  = (float*)alloc((size_t)S_LEN * KCH * 4);
  float* pdfg  = (float*)alloc((size_t)S_LEN * KCH * GQ * 4);
  float* lp    = (float*)alloc((size_t)S_LEN * KCH * GQ * 4);
  int*   selmap = (int*)alloc((size_t)S_LEN * KCH * 4);
  unsigned short* selbf = (unsigned short*)alloc((size_t)S_LEN * MSEL * HDIM * 2);
  float* cmean = (float*)alloc((size_t)S_LEN * HDIM * 4);
  unsigned short* hfbf  = (unsigned short*)alloc((size_t)S_LEN * HDIM * 2);
  float* qout  = (float*)alloc((size_t)S_LEN * HDIM * 4);
  float* kout  = (float*)alloc((size_t)S_LEN * MSEL * HDIM * 4);
  float* vout  = (float*)alloc((size_t)S_LEN * MSEL * HDIM * 4);
  unsigned short* hidbf = (unsigned short*)alloc((size_t)S_LEN * HDIM * 2);
  float* jsum  = (float*)alloc((size_t)S_LEN * 4);
  unsigned short* qwbf  = (unsigned short*)alloc((size_t)HDIM * HDIM * 2);
  unsigned short* kwbf  = (unsigned short*)alloc((size_t)HDIM * HDIM * 2);
  unsigned short* vwbf  = (unsigned short*)alloc((size_t)HDIM * HDIM * 2);
  unsigned short* lmbf  = (unsigned short*)alloc((size_t)VOCAB * HDIM * 2);

  float* logits = (float*)d_out;
  float* loss   = logits + (size_t)S_LEN * VOCAB;

  k_tables<<<1, 256, 0, stream>>>(lx, l1p, lnn, l1pn, glw);
  k_gather<<<(HDIM * S_LEN) / 256, 256, 0, stream>>>(ids, embed, baseT);
  k_cvt<<<(S_LEN * HDIM) / 256, 256, 0, stream>>>(h_final, hfbf, S_LEN * HDIM);
  k_cvt<<<(HDIM * HDIM) / 256, 256, 0, stream>>>(q_w, qwbf, HDIM * HDIM);
  k_cvt<<<(HDIM * HDIM) / 256, 256, 0, stream>>>(k_w, kwbf, HDIM * HDIM);
  k_cvt<<<(HDIM * HDIM) / 256, 256, 0, stream>>>(v_w, vwbf, HDIM * HDIM);
  k_cvt<<<(VOCAB * HDIM) / 256, 256, 0, stream>>>(lm_w, lmbf, VOCAB * HDIM);
  k_proj<<<(S_LEN * KCH) / 256, 256, 0, stream>>>(h_pen, focus_w, focus_b, imp_w, imp_b,
                                                  am1, bm1, cc, impv);
  k_pdfg<<<(S_LEN * KCH) / 256, 256, 0, stream>>>(am1, bm1, cc, lnn, l1pn, pdfg, lp);
  k_topk<<<S_LEN / 256, 256, 0, stream>>>(impv, selmap);
  k_js<<<S_LEN, 256, 0, stream>>>(pdfg, lp, glw, jsum);
  k_lossred<<<1, 256, 0, stream>>>(jsum, loss);
  k_chunk<<<S_LEN, 256, 0, stream>>>(am1, bm1, cc, impv, lx, l1p, selmap, baseT,
                                     selbf, cmean);
  k_gemm<<<S_LEN / 128, 256, 0, stream>>>(hfbf, qwbf, q_b, qout, S_LEN, HDIM);
  k_gemm<<<(S_LEN * MSEL) / 128, 256, 0, stream>>>(selbf, kwbf, k_b, kout, S_LEN * MSEL, HDIM);
  k_gemm<<<(S_LEN * MSEL) / 128, 256, 0, stream>>>(selbf, vwbf, v_b, vout, S_LEN * MSEL, HDIM);
  k_attn<<<S_LEN, 256, 0, stream>>>(qout, kout, vout, h_pen, h_final, cmean,
                                    gate_w, gate_b, ln_g, ln_b, hidbf);
  k_gemm<<<S_LEN / 128, 256, 0, stream>>>(hidbf, lmbf, nullptr, logits, S_LEN, VOCAB);
}
